// TopKMoE_69441031241775
// MI455X (gfx1250) — matrix operand order repaired and checked by reading, not compiled, not run on hardware
//
#include <hip/hip_runtime.h>
#include <hip/hip_bf16.h>
#include <stdint.h>

// ---------------------------------------------------------------------------
// Top-2 MoE (B=65536, D=256, E=16) for gfx1250 / MI455X.
// Route first, compute only the 2 selected experts per token with
// v_wmma_f32_16x16x32_bf16 (f32 accumulate). ~26 GFLOP instead of 146.
// 64-token tiles: each global B-fragment (weights) is reused across 4 M-tiles,
// cutting L2 weight traffic 4x vs 16-token tiles.
// ---------------------------------------------------------------------------

typedef __attribute__((ext_vector_type(16))) __bf16 v16bf;
typedef __attribute__((ext_vector_type(8)))  float  v8f;

union ABf { v16bf v; uint4 q[2]; };

#define DIM   256
#define NEXP  16
#define NTOK  65536
#define SLOPE 0.01f

static __device__ __forceinline__ __bf16 f2bf(float f) {
  // round-to-nearest-even f32 -> bf16
  uint32_t u = __float_as_uint(f);
  uint32_t r = (u + 0x7FFFu + ((u >> 16) & 1u)) >> 16;
  uint16_t s = (uint16_t)r;
  return __builtin_bit_cast(__bf16, s);
}

// A fragment (16x32 bf16) from a row-major [16][256] bf16 tile in LDS.
// lane L: M = L&15;  lanes 0-15 hold K = k0..k0+15,
//                    lanes 16-31 hold K = k0+16..k0+31.
static __device__ __forceinline__ v16bf load_a_lds(const __bf16* base, int lane, int k0) {
  int m    = lane & 15;
  int half = (lane >> 4) << 4;   // 0 or 16
  const __bf16* p = base + m * DIM + k0 + half;
  ABf u;
  u.q[0] = *(const uint4*)(p);
  u.q[1] = *(const uint4*)(p + 8);
  return u.v;
}

// B fragment (32x16 bf16) for C = X @ W^T with W row-major [O][256]:
// B[k][n] = W[n][k].  lane L: N = L&15; lanes 0-15 hold K=k0..k0+15,
// lanes 16-31 hold K=k0+16..k0+31  -> each lane reads 32 contiguous bytes
// of one W row.
static __device__ __forceinline__ v16bf load_b_w(const __bf16* wrow0, int lane, int k0) {
  int n    = lane & 15;
  int koff = k0 + ((lane >> 4) << 4);   // +0 or +16
  const __bf16* p = wrow0 + (size_t)n * DIM + koff;
  ABf u;
  u.q[0] = *(const uint4*)(p);
  u.q[1] = *(const uint4*)(p + 8);
  return u.v;
}

// ---------------------------------------------------------------------------
__global__ void cvt_f32_bf16(const float* __restrict__ s, __bf16* __restrict__ d, int n4) {
  int i = blockIdx.x * blockDim.x + threadIdx.x;
  if (i < n4) {
    float4 f = ((const float4*)s)[i];
    union { __bf16 b[4]; uint2 u; } o;
    o.b[0] = f2bf(f.x); o.b[1] = f2bf(f.y);
    o.b[2] = f2bf(f.z); o.b[3] = f2bf(f.w);
    ((uint2*)d)[i] = o.u;
  }
}

// ---------------------------------------------------------------------------
// Router: h = leaky(x@Wr1^T + br1); logits = h@Wr2^T + br2; top2 + softmax;
// bin tokens into per-expert lists.  One block = 64 tokens, 128 thr = 4 waves.
// Each wave owns 4 N-tiles; B fragment reused across the 4 M-tiles.
__global__ __launch_bounds__(128) void router_kernel(
    const __bf16* __restrict__ xb, const __bf16* __restrict__ Wr1b,
    const float* __restrict__ br1, const __bf16* __restrict__ Wr2b,
    const float* __restrict__ br2,
    int* __restrict__ counts, int* __restrict__ lists, float* __restrict__ probs)
{
  __shared__ __align__(16) __bf16 Xs[64 * DIM];   // 32 KB
  __shared__ __align__(16) __bf16 Hs[64 * DIM];   // 32 KB
  __shared__ float Ls[64 * 16];                   //  4 KB

  const int tid  = threadIdx.x;
  const int lane = tid & 31;
  const int w    = tid >> 5;
  const int t0   = blockIdx.x * 64;

  // stage 64 token rows (32 KB, fully contiguous) into LDS
  {
    const uint4* src = (const uint4*)(xb + (size_t)t0 * DIM);
    uint4* dst = (uint4*)Xs;
    #pragma unroll
    for (int it = 0; it < 16; ++it)
      dst[tid + it * 128] = src[tid + it * 128];
  }
  __syncthreads();

  // hidden layer: wave w owns N-tiles w*4..w*4+3, all 4 M-tiles each
  for (int i = 0; i < 4; ++i) {
    int n0 = (w * 4 + i) * 16;
    v8f acc[4] = {};
    #pragma unroll
    for (int kt = 0; kt < 8; ++kt) {
      int k0 = kt * 32;
      v16bf b = load_b_w(Wr1b + (size_t)n0 * DIM, lane, k0);
      #pragma unroll
      for (int mt = 0; mt < 4; ++mt) {
        v16bf a = load_a_lds(Xs + mt * 16 * DIM, lane, k0);
        acc[mt] = __builtin_amdgcn_wmma_f32_16x16x32_bf16(
            false, a, false, b, (short)0, acc[mt], false, false);
      }
    }
    int   N     = n0 + (lane & 15);
    float bias  = br1[N];
    int   mbase = (lane >> 4) * 8;
    #pragma unroll
    for (int mt = 0; mt < 4; ++mt) {
      #pragma unroll
      for (int v = 0; v < 8; ++v) {
        float h = acc[mt][v] + bias;
        h = h > 0.f ? h : SLOPE * h;
        Hs[(mt * 16 + mbase + v) * DIM + N] = f2bf(h);
      }
    }
  }
  __syncthreads();

  // logits (64x16): wave w computes M-tile w
  {
    v8f acc = {};
    #pragma unroll
    for (int kt = 0; kt < 8; ++kt) {
      int k0 = kt * 32;
      v16bf a = load_a_lds(Hs + w * 16 * DIM, lane, k0);
      v16bf b = load_b_w(Wr2b, lane, k0);
      acc = __builtin_amdgcn_wmma_f32_16x16x32_bf16(
          false, a, false, b, (short)0, acc, false, false);
    }
    int N = lane & 15;
    float bias = br2[N];
    int mbase = (lane >> 4) * 8;
    #pragma unroll
    for (int v = 0; v < 8; ++v)
      Ls[(w * 16 + mbase + v) * 16 + N] = acc[v] + bias;
  }
  __syncthreads();

  // top-2 + softmax + expert binning; thread t handles token t (t < 64)
  if (tid < 64) {
    const float* l = Ls + tid * 16;
    float b0 = -1e30f, b1 = -1e30f; int i0 = 0, i1 = 0;
    #pragma unroll
    for (int j = 0; j < NEXP; ++j) {
      float v = l[j];
      if (v > b0)      { b1 = b0; i1 = i0; b0 = v; i0 = j; }
      else if (v > b1) { b1 = v;  i1 = j; }
    }
    float e1  = __expf(b1 - b0);
    float inv = 1.f / (1.f + e1);
    int tg = t0 + tid;
    probs[tg * 2 + 0] = inv;
    probs[tg * 2 + 1] = e1 * inv;
    int p0 = atomicAdd(&counts[i0], 1);
    lists[i0 * NTOK + p0] = (tg << 1);
    int p1 = atomicAdd(&counts[i1], 1);
    lists[i1 * NTOK + p1] = (tg << 1) | 1;
  }
}

// ---------------------------------------------------------------------------
// Expert compute: out[tok] += p * (We[e] @ x[tok] + be[e]).
// grid.y = expert, grid.x = 64-token tile of that expert's list.
__global__ __launch_bounds__(128) void expert_kernel(
    const __bf16* __restrict__ xb, const __bf16* __restrict__ Web,
    const float* __restrict__ be,
    const int* __restrict__ counts, const int* __restrict__ lists,
    const float* __restrict__ probs, float* __restrict__ out)
{
  const int e     = blockIdx.y;
  const int cnt   = counts[e];
  const int start = blockIdx.x * 64;
  if (start >= cnt) return;
  const int rem = cnt - start;
  const int nv  = rem < 64 ? rem : 64;

  __shared__ __align__(16) __bf16 Xs[64 * DIM];   // 32 KB
  __shared__ int   tok[64];
  __shared__ float pw[64];

  const int tid  = threadIdx.x;
  const int lane = tid & 31;
  const int w    = tid >> 5;

  if (tid < 64) {
    if (tid < nv) {
      int ent = lists[e * NTOK + start + tid];
      tok[tid] = ent >> 1;
      pw[tid]  = probs[ent];
    } else { tok[tid] = 0; pw[tid] = 0.f; }
  }
  __syncthreads();

  // gather 64 token rows; thread pair per row (256 B each half)
  {
    int r = tid >> 1;
    int c = (tid & 1) * 128;            // in bf16 elements
    uint4* dst = (uint4*)(Xs + r * DIM + c);
    if (r < nv) {
      const uint4* src = (const uint4*)(xb + (size_t)tok[r] * DIM + c);
      #pragma unroll
      for (int j = 0; j < 16; ++j) dst[j] = src[j];
    } else {
      uint4 z = {0u, 0u, 0u, 0u};
      #pragma unroll
      for (int j = 0; j < 16; ++j) dst[j] = z;
    }
  }
  __syncthreads();

  const __bf16* We_e = Web + (size_t)e * DIM * DIM;
  for (int i = 0; i < 4; ++i) {
    int n0 = (w * 4 + i) * 16;
    v8f acc[4] = {};
    #pragma unroll
    for (int kt = 0; kt < 8; ++kt) {
      int k0 = kt * 32;
      v16bf b = load_b_w(We_e + (size_t)n0 * DIM, lane, k0);
      #pragma unroll
      for (int mt = 0; mt < 4; ++mt) {
        v16bf a = load_a_lds(Xs + mt * 16 * DIM, lane, k0);
        acc[mt] = __builtin_amdgcn_wmma_f32_16x16x32_bf16(
            false, a, false, b, (short)0, acc[mt], false, false);
      }
    }
    int   N     = n0 + (lane & 15);
    float bias  = be[e * DIM + N];
    int   mbase = (lane >> 4) * 8;
    #pragma unroll
    for (int mt = 0; mt < 4; ++mt) {
      #pragma unroll
      for (int v = 0; v < 8; ++v) {
        int m = mt * 16 + mbase + v;
        if (m < nv) {
          float val = pw[m] * (acc[mt][v] + bias);
          atomicAdd(&out[(size_t)tok[m] * DIM + N], val);  // exactly 2 adds/elem
        }
      }
    }
  }
}

// ---------------------------------------------------------------------------
__global__ void leaky_kernel(float* __restrict__ o, int n) {
  int i = blockIdx.x * blockDim.x + threadIdx.x;
  if (i < n) { float v = o[i]; o[i] = v > 0.f ? v : SLOPE * v; }
}

// ---------------------------------------------------------------------------
extern "C" void kernel_launch(void* const* d_in, const int* in_sizes, int n_in,
                              void* d_out, int out_size, void* d_ws, size_t ws_size,
                              hipStream_t stream)
{
  (void)in_sizes; (void)n_in; (void)ws_size;
  const float* x   = (const float*)d_in[0];
  const float* Wr1 = (const float*)d_in[1];
  const float* br1 = (const float*)d_in[2];
  const float* Wr2 = (const float*)d_in[3];
  const float* br2 = (const float*)d_in[4];
  const float* We  = (const float*)d_in[5];
  const float* be  = (const float*)d_in[6];
  float* out = (float*)d_out;

  // workspace layout (bytes), all offsets 256-aligned; total ~38.7 MB
  char* ws = (char*)d_ws;
  __bf16* xb     = (__bf16*)(ws);                 // 65536*256*2  = 32 MB
  __bf16* Web    = (__bf16*)(ws + 33554432);      // 16*256*256*2 =  2 MB
  __bf16* Wr1b   = (__bf16*)(ws + 35651584);      // 256*256*2    = 128 KB
  __bf16* Wr2b   = (__bf16*)(ws + 35782656);      // 16*256*2     =   8 KB
  int*    counts = (int*)   (ws + 35790848);      // 16 ints (pad 256)
  int*    lists  = (int*)   (ws + 35791104);      // 16*65536*4   =  4 MB
  float*  probs  = (float*) (ws + 39985408);      // 65536*2*4    = 512 KB

  hipMemsetAsync(counts, 0, 256, stream);
  hipMemsetAsync(d_out, 0, (size_t)out_size * sizeof(float), stream);

  // f32 -> bf16 conversions
  {
    int n4;
    n4 = (NTOK * DIM) / 4;
    cvt_f32_bf16<<<(n4 + 255) / 256, 256, 0, stream>>>(x, xb, n4);
    n4 = (NEXP * DIM * DIM) / 4;
    cvt_f32_bf16<<<(n4 + 255) / 256, 256, 0, stream>>>(We, Web, n4);
    n4 = (DIM * DIM) / 4;
    cvt_f32_bf16<<<(n4 + 255) / 256, 256, 0, stream>>>(Wr1, Wr1b, n4);
    n4 = (NEXP * DIM) / 4;
    cvt_f32_bf16<<<(n4 + 255) / 256, 256, 0, stream>>>(Wr2, Wr2b, n4);
  }

  router_kernel<<<NTOK / 64, 128, 0, stream>>>(xb, Wr1b, br1, Wr2b, br2,
                                               counts, lists, probs);
  expert_kernel<<<dim3(NTOK / 64, NEXP, 1), 128, 0, stream>>>(
      xb, Web, be, counts, lists, probs, out);
  leaky_kernel<<<(out_size + 255) / 256, 256, 0, stream>>>(out, out_size);
}
